// StructureModule_46145128628301
// MI455X (gfx1250) — matrix operand order repaired and checked by reading, not compiled, not run on hardware
//
#include <hip/hip_runtime.h>
#include <math.h>

// ---------------------------------------------------------------------------
// AlphaFold-style StructureModule forward for MI455X (gfx1250, wave32, WMMA).
// All heavy GEMMs run through one generic batched bf16 WMMA kernel
// (v_wmma_f32_16x16x32_bf16), accumulating in f32. The pair tensor is
// normalized once into bf16 (151 MB -> fits the 192 MB L2) and reused by both
// pair-consuming GEMMs each layer. Interior tiles are staged to LDS with the
// CDNA5 async path (global_load_async_to_lds_b128 + s_wait_asynccnt).
// ---------------------------------------------------------------------------

#ifndef USE_ASYNC_LDS
#define USE_ASYNC_LDS 1
#endif

typedef __bf16 bf16_t;
typedef __attribute__((ext_vector_type(16))) __bf16 v16bf;
typedef __attribute__((ext_vector_type(8)))  __bf16 v8bf;
typedef __attribute__((ext_vector_type(8)))  float  v8f;

#define WLc 0.57735026919f   /* sqrt(1/3) */
#define WCc 0.23570226039f   /* sqrt(2/(9*4)) */

// ---------------------------------------------------------------------------
// Generic batched GEMM:  C[M x N] = act( A[M x K] * B + bias + residual )
// A: bf16 row-major (lda, batchA).  B: bf16, either normal [K x N] (ldb = row
// stride) or pre-transposed BT [N x K].  Output optionally f32 and/or bf16.
// Block tile 64x64, 128 threads = 4 waves, each wave 2x2 WMMA 16x16x32 tiles.
// ---------------------------------------------------------------------------
struct GemmArgs {
  const bf16_t* A; long long lda; long long batchA;
  const bf16_t* B; long long ldb; long long batchB; int btr;
  float* Cf; bf16_t* Cb; long long ldc; long long batchC;
  const float* bias;
  const float* residual; long long ldres; long long batchRes;
  int M, N, K, relu;
};

__launch_bounds__(128)
__global__ void gemm_bf16_wmma(GemmArgs g) {
  const int tid  = threadIdx.x;
  const int wave = tid >> 5;
  const int lane = tid & 31;
  const int wm = wave >> 1, wn = wave & 1;
  const int m0 = blockIdx.y * 64, n0 = blockIdx.x * 64;
  const int bz = blockIdx.z;

  const bf16_t* A = g.A + (long long)bz * g.batchA;
  const bf16_t* B = g.B + (long long)bz * g.batchB;

  __shared__ __align__(16) __bf16 As[64 * 40];
  __shared__ __align__(16) __bf16 Bs[64 * 40];   // stored as BT: [n][k]

  v8bf zero8;
#pragma unroll
  for (int e = 0; e < 8; ++e) zero8[e] = (__bf16)0.0f;

  v8f acc[2][2];
#pragma unroll
  for (int ti = 0; ti < 2; ++ti)
#pragma unroll
    for (int tj = 0; tj < 2; ++tj)
#pragma unroll
      for (int e = 0; e < 8; ++e) acc[ti][tj][e] = 0.0f;

  const int mlane = lane & 15;
  const int kbA = (lane < 16) ? 0 : 8;    // A frag: K kb..kb+7 and kb+16..kb+23
  const int kbB = (lane < 16) ? 0 : 8;    // B frag: K kb..kb+7 and kb+16..kb+23

  const bool fullA = (m0 + 64 <= g.M);
  const bool fullB = (n0 + 64 <= g.N);

  for (int k0 = 0; k0 < g.K; k0 += 32) {
    const bool fullK = (k0 + 32 <= g.K);
    int asyncs = 0;
    // ---- stage A tile (64 x 32) ----
#if USE_ASYNC_LDS
    if (fullA && fullK) {
#pragma unroll
      for (int it = 0; it < 2; ++it) {
        int chunk = it * 128 + tid;          // 256 chunks of 8 elements
        int r = chunk >> 2, c = (chunk & 3) * 8;
        const bf16_t* gp = A + (long long)(m0 + r) * g.lda + (k0 + c);
        // low 32 bits of a generic shared pointer == LDS byte offset
        unsigned lds_off = (unsigned)(unsigned long long)(&As[r * 40 + c]);
        asm volatile("global_load_async_to_lds_b128 %0, %1, off"
                     :: "v"(lds_off), "v"(gp) : "memory");
      }
      asyncs = 1;
    } else
#endif
    {
#pragma unroll
      for (int it = 0; it < 2; ++it) {
        int chunk = it * 128 + tid;
        int r = chunk >> 2, c = (chunk & 3) * 8;
        v8bf val = zero8;
        int gm = m0 + r, gk = k0 + c;
        if (gm < g.M && gk < g.K)
          val = *(const v8bf*)(A + (long long)gm * g.lda + gk);
        *(v8bf*)(&As[r * 40 + c]) = val;
      }
    }
    // ---- stage B tile as BT (64 x 32) ----
    if (g.btr) {
#if USE_ASYNC_LDS
      if (fullB && fullK) {
#pragma unroll
        for (int it = 0; it < 2; ++it) {
          int chunk = it * 128 + tid;
          int r = chunk >> 2, c = (chunk & 3) * 8;   // r = n, c = k
          const bf16_t* gp = B + (long long)(n0 + r) * g.ldb + (k0 + c);
          unsigned lds_off = (unsigned)(unsigned long long)(&Bs[r * 40 + c]);
          asm volatile("global_load_async_to_lds_b128 %0, %1, off"
                       :: "v"(lds_off), "v"(gp) : "memory");
        }
        asyncs = 1;
      } else
#endif
      {
#pragma unroll
        for (int it = 0; it < 2; ++it) {
          int chunk = it * 128 + tid;
          int r = chunk >> 2, c = (chunk & 3) * 8;
          v8bf val = zero8;
          int gn = n0 + r, gk = k0 + c;
          if (gn < g.N && gk < g.K)
            val = *(const v8bf*)(B + (long long)gn * g.ldb + gk);
          *(v8bf*)(&Bs[r * 40 + c]) = val;
        }
      }
    } else {
#pragma unroll
      for (int it = 0; it < 2; ++it) {
        int chunk = it * 128 + tid;        // 256 chunks: k in [0,32), 8 n's each
        int kk = chunk >> 3, nn = (chunk & 7) * 8;
        int gk = k0 + kk;
#pragma unroll
        for (int j = 0; j < 8; ++j) {
          int gn = n0 + nn + j;
          __bf16 v = (__bf16)0.0f;
          if (gk < g.K && gn < g.N) v = B[(long long)gk * g.ldb + gn];
          Bs[(nn + j) * 40 + kk] = v;
        }
      }
    }
#if USE_ASYNC_LDS
    if (asyncs) asm volatile("s_wait_asynccnt 0x0" ::: "memory");
#endif
    __syncthreads();

    // prefetch next A tile (global_prefetch_b8 on gfx1250)
    if (k0 + 32 < g.K) {
      int gm = m0 + (tid >> 2), gk = k0 + 32 + (tid & 3) * 8;
      if (gm < g.M && gk < g.K)
        __builtin_prefetch(A + (long long)gm * g.lda + gk, 0, 1);
    }

    // ---- fragments + WMMA ----
    v16bf afrag[2], bfrag[2];
#pragma unroll
    for (int ti = 0; ti < 2; ++ti) {
      const __bf16* p = &As[(wm * 32 + ti * 16 + mlane) * 40 + kbA];
      v8bf lo = *(const v8bf*)p;
      v8bf hi = *(const v8bf*)(p + 16);
#pragma unroll
      for (int e = 0; e < 8; ++e) { afrag[ti][e] = lo[e]; afrag[ti][8 + e] = hi[e]; }
    }
#pragma unroll
    for (int tj = 0; tj < 2; ++tj) {
      const __bf16* p = &Bs[(wn * 32 + tj * 16 + mlane) * 40 + kbB];
      v8bf lo = *(const v8bf*)p;
      v8bf hi = *(const v8bf*)(p + 16);
#pragma unroll
      for (int e = 0; e < 8; ++e) { bfrag[tj][e] = lo[e]; bfrag[tj][8 + e] = hi[e]; }
    }
#pragma unroll
    for (int ti = 0; ti < 2; ++ti)
#pragma unroll
      for (int tj = 0; tj < 2; ++tj)
        acc[ti][tj] = __builtin_amdgcn_wmma_f32_16x16x32_bf16(
            false, afrag[ti], false, bfrag[tj], (short)0, acc[ti][tj], false, false);
    __syncthreads();
  }

  // ---- epilogue: lanes 0-15 -> rows 0..7, lanes 16-31 -> rows 8..15 ----
  const int colB  = lane & 15;
  const int rbase = (lane < 16) ? 0 : 8;
#pragma unroll
  for (int ti = 0; ti < 2; ++ti)
#pragma unroll
    for (int tj = 0; tj < 2; ++tj)
#pragma unroll
      for (int j = 0; j < 8; ++j) {
        int m = m0 + wm * 32 + ti * 16 + rbase + j;
        int n = n0 + wn * 32 + tj * 16 + colB;
        if (m < g.M && n < g.N) {
          float v = acc[ti][tj][j];
          if (g.bias)     v += g.bias[n];
          if (g.residual) v += g.residual[(long long)bz * g.batchRes + (long long)m * g.ldres + n];
          if (g.relu && v < 0.0f) v = 0.0f;
          long long off = (long long)bz * g.batchC + (long long)m * g.ldc + n;
          if (g.Cf) g.Cf[off] = v;
          if (g.Cb) g.Cb[off] = (__bf16)v;
        }
      }
}

// ---------------------------------------------------------------------------
// f32 -> bf16 conversion (weights arena)
// ---------------------------------------------------------------------------
__launch_bounds__(256)
__global__ void f32_to_bf16(const float* __restrict__ src, bf16_t* __restrict__ dst,
                            long long count) {
  long long idx = (long long)blockIdx.x * 256 + threadIdx.x;
  if (idx < count) dst[idx] = (__bf16)src[idx];
}

// ---------------------------------------------------------------------------
// LayerNorm over rows (one block per row, 128 threads)
// ---------------------------------------------------------------------------
__launch_bounds__(128)
__global__ void ln_rows(const float* __restrict__ x, const float* __restrict__ gw,
                        const float* __restrict__ bw, float* __restrict__ yf,
                        bf16_t* __restrict__ yb, int W) {
  const long long row = blockIdx.x;
  const float* xr = x + row * W;
  const int tid = threadIdx.x;
  __shared__ float red[128];
  float s = 0.0f;
  for (int c = tid; c < W; c += 128) s += xr[c];
  red[tid] = s; __syncthreads();
  for (int off = 64; off; off >>= 1) { if (tid < off) red[tid] += red[tid + off]; __syncthreads(); }
  const float mu = red[0] / W; __syncthreads();
  float v = 0.0f;
  for (int c = tid; c < W; c += 128) { float d = xr[c] - mu; v += d * d; }
  red[tid] = v; __syncthreads();
  for (int off = 64; off; off >>= 1) { if (tid < off) red[tid] += red[tid + off]; __syncthreads(); }
  const float rstd = rsqrtf(red[0] / W + 1e-5f);
  for (int c = tid; c < W; c += 128) {
    float o = (xr[c] - mu) * rstd * gw[c] + bw[c];
    if (yf) yf[row * W + c] = o;
    if (yb) yb[row * W + c] = (__bf16)o;
  }
}

// ---------------------------------------------------------------------------
// Build augmented attention operands per (n, h): one wave per block.
// A-hat[0:16]=q/4, A-hat[16:28]=sqrt(g*WC)*(R qp + t); same for B-hat with k.
// Also Vcat = [v(16) | R vp + t (24)] and the -0.5 g WC |.|^2 rank-1 terms.
// ---------------------------------------------------------------------------
__launch_bounds__(32)
__global__ void build_attn_inputs(const float* __restrict__ q, const float* __restrict__ k,
                                  const float* __restrict__ v, const float* __restrict__ qp,
                                  const float* __restrict__ kp, const float* __restrict__ vp,
                                  const float* __restrict__ Rb, const float* __restrict__ tb,
                                  const float* __restrict__ gamma,
                                  bf16_t* __restrict__ Ab, bf16_t* __restrict__ Bb,
                                  bf16_t* __restrict__ Vc, float* __restrict__ rowterm,
                                  float* __restrict__ colterm) {
  const int n = blockIdx.x, h = blockIdx.y, lane = threadIdx.x;
  float Rm[9], t3[3];
#pragma unroll
  for (int j = 0; j < 9; ++j) Rm[j] = Rb[n * 9 + j];
#pragma unroll
  for (int j = 0; j < 3; ++j) t3[j] = tb[n * 3 + j];
  const float gx = gamma[h];
  const float g  = (gx > 20.0f) ? gx : log1pf(__expf(gx));
  const float sg = sqrtf(g * WCc);
  __shared__ float red[32];

  // A-hat + rowterm
  float aval = 0.0f, q2 = 0.0f;
  if (lane < 16) {
    aval = q[((long long)n * 12 + h) * 16 + lane] * 0.25f;   // 1/sqrt(C)
  } else if (lane < 28) {
    int j = lane - 16, p = j / 3, i = j % 3;
    const float* qpp = qp + ((long long)n * 12 + h) * 12 + p * 3;
    float gq = Rm[i * 3] * qpp[0] + Rm[i * 3 + 1] * qpp[1] + Rm[i * 3 + 2] * qpp[2] + t3[i];
    q2 = gq * gq;
    aval = sg * gq;
  }
  Ab[((long long)h * 768 + n) * 32 + lane] = (__bf16)aval;
  red[lane] = q2; __syncthreads();
  if (lane == 0) {
    float s = 0.0f;
    for (int j = 16; j < 28; ++j) s += red[j];
    rowterm[h * 768 + n] = -0.5f * g * WCc * s;
  }
  __syncthreads();

  // B-hat + colterm
  float bval = 0.0f, k2 = 0.0f;
  if (lane < 16) {
    bval = k[((long long)n * 12 + h) * 16 + lane];
  } else if (lane < 28) {
    int j = lane - 16, p = j / 3, i = j % 3;
    const float* kpp = kp + ((long long)n * 12 + h) * 12 + p * 3;
    float gk = Rm[i * 3] * kpp[0] + Rm[i * 3 + 1] * kpp[1] + Rm[i * 3 + 2] * kpp[2] + t3[i];
    k2 = gk * gk;
    bval = sg * gk;
  }
  Bb[((long long)h * 768 + n) * 32 + lane] = (__bf16)bval;
  red[lane] = k2; __syncthreads();
  if (lane == 0) {
    float s = 0.0f;
    for (int j = 16; j < 28; ++j) s += red[j];
    colterm[h * 768 + n] = -0.5f * g * WCc * s;
  }

  // Vcat = [v | vpg]
  if (lane < 16)
    Vc[((long long)n * 12 + h) * 40 + lane] = (__bf16)v[((long long)n * 12 + h) * 16 + lane];
  if (lane < 24) {
    int p = lane / 3, i = lane % 3;
    const float* vpp = vp + ((long long)n * 12 + h) * 24 + p * 3;
    float gv = Rm[i * 3] * vpp[0] + Rm[i * 3 + 1] * vpp[1] + Rm[i * 3 + 2] * vpp[2] + t3[i];
    Vc[((long long)n * 12 + h) * 40 + 16 + lane] = (__bf16)gv;
  }
}

// ---------------------------------------------------------------------------
// Softmax over m for each (h, n); adds pair bias + rank-1 distance terms.
// ---------------------------------------------------------------------------
__launch_bounds__(256)
__global__ void softmax_rows(const float* __restrict__ logits, const float* __restrict__ biasbuf,
                             const float* __restrict__ rowterm, const float* __restrict__ colterm,
                             bf16_t* __restrict__ attn) {
  const int n = blockIdx.x, h = blockIdx.y, tid = threadIdx.x;
  const float* lr = logits + ((long long)h * 768 + n) * 768;
  const float rt = rowterm[h * 768 + n];
  __shared__ float red[256];
  float vals[3], mx = -3.0e38f;
#pragma unroll
  for (int j = 0; j < 3; ++j) {
    int m = tid + j * 256;
    float vv = WLc * (lr[m] + biasbuf[((long long)n * 768 + m) * 16 + h] + rt + colterm[h * 768 + m]);
    vals[j] = vv; mx = fmaxf(mx, vv);
  }
  red[tid] = mx; __syncthreads();
  for (int off = 128; off; off >>= 1) { if (tid < off) red[tid] = fmaxf(red[tid], red[tid + off]); __syncthreads(); }
  mx = red[0]; __syncthreads();
  float s = 0.0f;
#pragma unroll
  for (int j = 0; j < 3; ++j) { vals[j] = __expf(vals[j] - mx); s += vals[j]; }
  red[tid] = s; __syncthreads();
  for (int off = 128; off; off >>= 1) { if (tid < off) red[tid] += red[tid + off]; __syncthreads(); }
  const float inv = 1.0f / red[0];
  bf16_t* ar = attn + ((long long)h * 768 + n) * 768;
#pragma unroll
  for (int j = 0; j < 3; ++j) ar[tid + j * 256] = (__bf16)(vals[j] * inv);
}

// ---------------------------------------------------------------------------
// Build cat = [o | R^T(o_ptg - t) | ||o_pt|| | o_pair], bf16, per n.
// ---------------------------------------------------------------------------
__launch_bounds__(256)
__global__ void build_cat(const float* __restrict__ Ocat, const float* __restrict__ opair,
                          const float* __restrict__ Rb, const float* __restrict__ tb,
                          bf16_t* __restrict__ cat) {
  const int n = blockIdx.x, tid = threadIdx.x;
  bf16_t* cn = cat + (long long)n * 2112;
  const float* On = Ocat + (long long)n * 480;
  for (int idx = tid; idx < 192; idx += 256) {
    int h = idx >> 4, c = idx & 15;
    cn[idx] = (__bf16)On[h * 40 + c];
  }
  if (tid < 96) {
    int h = tid >> 3, p = tid & 7;
    float Rm[9], t3[3];
#pragma unroll
    for (int j = 0; j < 9; ++j) Rm[j] = Rb[n * 9 + j];
#pragma unroll
    for (int j = 0; j < 3; ++j) t3[j] = tb[n * 3 + j];
    float og[3], s2 = 1e-8f;
#pragma unroll
    for (int i = 0; i < 3; ++i) og[i] = On[h * 40 + 16 + p * 3 + i] - t3[i];
#pragma unroll
    for (int i = 0; i < 3; ++i) {
      float opt = Rm[0 + i] * og[0] + Rm[3 + i] * og[1] + Rm[6 + i] * og[2];  // R^T
      s2 += opt * opt;
      cn[192 + h * 24 + p * 3 + i] = (__bf16)opt;
    }
    cn[480 + h * 8 + p] = (__bf16)sqrtf(s2);
  }
  for (int idx = tid; idx < 1536; idx += 256) {
    cn[576 + idx] = (__bf16)opair[(long long)n * 1536 + idx];
  }
}

// ---------------------------------------------------------------------------
// bu = S @ wbu + bbu  ->  quaternion frame update (per n)
// ---------------------------------------------------------------------------
__launch_bounds__(64)
__global__ void frame_update(const float* __restrict__ S, const float* __restrict__ wbu,
                             const float* __restrict__ bbu, float* Rb, float* tb) {
  const int n = blockIdx.x, tid = threadIdx.x;
  __shared__ float red[64];
  __shared__ float bu[6];
  float acc[6] = {0, 0, 0, 0, 0, 0};
  for (int c = tid; c < 384; c += 64) {
    float s = S[(long long)n * 384 + c];
#pragma unroll
    for (int o = 0; o < 6; ++o) acc[o] += s * wbu[c * 6 + o];
  }
  for (int o = 0; o < 6; ++o) {
    red[tid] = acc[o]; __syncthreads();
    for (int off = 32; off; off >>= 1) { if (tid < off) red[tid] += red[tid + off]; __syncthreads(); }
    if (tid == 0) bu[o] = red[0] + bbu[o];
    __syncthreads();
  }
  if (tid == 0) {
    float b = bu[0], c = bu[1], d = bu[2], a = 1.0f;
    float inv = rsqrtf(a * a + b * b + c * c + d * d);
    a *= inv; b *= inv; c *= inv; d *= inv;
    float Ru[9] = {
      a * a + b * b - c * c - d * d, 2 * (b * c - a * d),          2 * (b * d + a * c),
      2 * (b * c + a * d),           a * a - b * b + c * c - d * d, 2 * (c * d - a * b),
      2 * (b * d - a * c),           2 * (c * d + a * b),           a * a - b * b - c * c + d * d };
    float Ro[9], to[3], tu[3] = {bu[3], bu[4], bu[5]}, Rn[9], tn[3];
#pragma unroll
    for (int j = 0; j < 9; ++j) Ro[j] = Rb[n * 9 + j];
#pragma unroll
    for (int j = 0; j < 3; ++j) to[j] = tb[n * 3 + j];
#pragma unroll
    for (int i = 0; i < 3; ++i) {
      tn[i] = Ro[i * 3] * tu[0] + Ro[i * 3 + 1] * tu[1] + Ro[i * 3 + 2] * tu[2] + to[i];
#pragma unroll
      for (int kk = 0; kk < 3; ++kk)
        Rn[i * 3 + kk] = Ro[i * 3] * Ru[kk] + Ro[i * 3 + 1] * Ru[3 + kk] + Ro[i * 3 + 2] * Ru[6 + kk];
    }
#pragma unroll
    for (int j = 0; j < 9; ++j) Rb[n * 9 + j] = Rn[j];
#pragma unroll
    for (int j = 0; j < 3; ++j) tb[n * 3 + j] = tn[j];
  }
}

__launch_bounds__(32)
__global__ void init_frames(float* Rb, float* tb) {
  const int n = blockIdx.x, tid = threadIdx.x;
  if (tid < 9) Rb[n * 9 + tid] = (tid == 0 || tid == 4 || tid == 8) ? 1.0f : 0.0f;
  if (tid < 3) tb[n * 3 + tid] = 0.0f;
}

__launch_bounds__(256)
__global__ void final_coords(const float* __restrict__ local, const float* __restrict__ Rb,
                             const float* __restrict__ tb, float* __restrict__ out) {
  int idx = blockIdx.x * 256 + threadIdx.x;   // over 768*3 points
  if (idx >= 768 * 3) return;
  int n = idx / 3, a = idx % 3;
  const float* lc = local + (long long)n * 9 + a * 3;
#pragma unroll
  for (int i = 0; i < 3; ++i)
    out[(long long)n * 9 + a * 3 + i] =
        Rb[n * 9 + i * 3] * lc[0] + Rb[n * 9 + i * 3 + 1] * lc[1] +
        Rb[n * 9 + i * 3 + 2] * lc[2] + tb[n * 3 + i];
}

// ---------------------------------------------------------------------------
// Host orchestration
// ---------------------------------------------------------------------------
extern "C" void kernel_launch(void* const* d_in, const int* in_sizes, int n_in,
                              void* d_out, int out_size, void* d_ws, size_t ws_size,
                              hipStream_t stream) {
  (void)in_sizes; (void)n_in; (void)out_size; (void)ws_size;

  // jax tree-leaf order (dicts sorted by key):
  const float* LOCAL = (const float*)d_in[0];   // local_coords [768,3,3]
  const float* PAIR  = (const float*)d_in[1];   // pair [768,768,128]
  const float* BSv   = (const float*)d_in[2];   // params.bs
  const float* B1    = (const float*)d_in[3];
  const float* B2    = (const float*)d_in[4];
  const float* B3    = (const float*)d_in[5];
  const float* BBU   = (const float*)d_in[6];
  const float* BO    = (const float*)d_in[7];
  const float* GAMMA = (const float*)d_in[8];
  const float* LN1B  = (const float*)d_in[9];
  const float* LN1G  = (const float*)d_in[10];
  const float* LN2B  = (const float*)d_in[11];
  const float* LN2G  = (const float*)d_in[12];
  const float* W1    = (const float*)d_in[13];
  const float* W2    = (const float*)d_in[14];
  const float* W3    = (const float*)d_in[15];
  const float* WB    = (const float*)d_in[16];
  const float* WBU   = (const float*)d_in[17];
  const float* WK    = (const float*)d_in[18];
  const float* WKP   = (const float*)d_in[19];
  const float* WO    = (const float*)d_in[20];
  const float* WQ    = (const float*)d_in[21];
  const float* WQP   = (const float*)d_in[22];
  const float* WV    = (const float*)d_in[23];
  const float* WVP   = (const float*)d_in[24];
  const float* PLNB  = (const float*)d_in[25];
  const float* PLNG  = (const float*)d_in[26];
  const float* SLNB  = (const float*)d_in[27];
  const float* SLNG  = (const float*)d_in[28];
  const float* WS    = (const float*)d_in[29];
  const float* SINGLE= (const float*)d_in[30];

  // workspace carve-out (256B aligned)
  char* wsp = (char*)d_ws;
  auto alloc = [&](size_t bytes) -> void* {
    void* p = wsp; wsp += (bytes + 255) & ~(size_t)255; return p;
  };
  bf16_t* pairb   = (bf16_t*)alloc((size_t)768 * 768 * 128 * 2);
  float*  biasbuf = (float*) alloc((size_t)768 * 768 * 16 * 4);
  float*  logits  = (float*) alloc((size_t)12 * 768 * 768 * 4);
  bf16_t* attnb   = (bf16_t*)alloc((size_t)12 * 768 * 768 * 2);
  float*  qf  = (float*)alloc((size_t)768 * 192 * 4);
  float*  kf  = (float*)alloc((size_t)768 * 192 * 4);
  float*  vf  = (float*)alloc((size_t)768 * 192 * 4);
  float*  qpf = (float*)alloc((size_t)768 * 144 * 4);
  float*  kpf = (float*)alloc((size_t)768 * 144 * 4);
  float*  vpf = (float*)alloc((size_t)768 * 288 * 4);
  bf16_t* Abuf = (bf16_t*)alloc((size_t)12 * 768 * 32 * 2);
  bf16_t* Bbuf = (bf16_t*)alloc((size_t)12 * 768 * 32 * 2);
  bf16_t* Vcat = (bf16_t*)alloc((size_t)768 * 480 * 2);
  float*  rowterm = (float*)alloc((size_t)12 * 768 * 4);
  float*  colterm = (float*)alloc((size_t)12 * 768 * 4);
  float*  Ocat  = (float*)alloc((size_t)768 * 480 * 4);
  float*  opair = (float*)alloc((size_t)768 * 1536 * 4);
  bf16_t* cat   = (bf16_t*)alloc((size_t)768 * 2112 * 2);
  bf16_t* Sh0   = (bf16_t*)alloc((size_t)768 * 384 * 2);
  float*  Sf    = (float*) alloc((size_t)768 * 384 * 4);
  bf16_t* Sbf   = (bf16_t*)alloc((size_t)768 * 384 * 2);
  float*  Xf    = (float*) alloc((size_t)768 * 384 * 4);
  float*  Yf    = (float*) alloc((size_t)768 * 384 * 4);
  bf16_t* Ybf   = (bf16_t*)alloc((size_t)768 * 384 * 2);
  float*  Zf    = (float*) alloc((size_t)768 * 384 * 4);
  bf16_t* h1    = (bf16_t*)alloc((size_t)768 * 384 * 2);
  bf16_t* h2    = (bf16_t*)alloc((size_t)768 * 384 * 2);
  float*  Rbuf  = (float*) alloc((size_t)768 * 9 * 4);
  float*  tbuf  = (float*) alloc((size_t)768 * 3 * 4);

  // bf16 weight arena: convert every stacked param array once per call
  bf16_t* W1b  = (bf16_t*)alloc((size_t)8 * 384 * 384 * 2);
  bf16_t* W2b  = (bf16_t*)alloc((size_t)8 * 384 * 384 * 2);
  bf16_t* W3b  = (bf16_t*)alloc((size_t)8 * 384 * 384 * 2);
  bf16_t* WQb  = (bf16_t*)alloc((size_t)8 * 384 * 192 * 2);
  bf16_t* WKb  = (bf16_t*)alloc((size_t)8 * 384 * 192 * 2);
  bf16_t* WVb  = (bf16_t*)alloc((size_t)8 * 384 * 192 * 2);
  bf16_t* WQPb = (bf16_t*)alloc((size_t)8 * 384 * 144 * 2);
  bf16_t* WKPb = (bf16_t*)alloc((size_t)8 * 384 * 144 * 2);
  bf16_t* WVPb = (bf16_t*)alloc((size_t)8 * 384 * 288 * 2);
  bf16_t* WBb  = (bf16_t*)alloc((size_t)8 * 128 * 12 * 2);
  bf16_t* WOb  = (bf16_t*)alloc((size_t)8 * 2112 * 384 * 2);
  bf16_t* WSb  = (bf16_t*)alloc((size_t)384 * 384 * 2);

  auto convert = [&](const float* src, bf16_t* dst, size_t count) {
    f32_to_bf16<<<(int)((count + 255) / 256), 256, 0, stream>>>(src, dst, (long long)count);
  };
  convert(W1,  W1b,  (size_t)8 * 384 * 384);
  convert(W2,  W2b,  (size_t)8 * 384 * 384);
  convert(W3,  W3b,  (size_t)8 * 384 * 384);
  convert(WQ,  WQb,  (size_t)8 * 384 * 192);
  convert(WK,  WKb,  (size_t)8 * 384 * 192);
  convert(WV,  WVb,  (size_t)8 * 384 * 192);
  convert(WQP, WQPb, (size_t)8 * 384 * 144);
  convert(WKP, WKPb, (size_t)8 * 384 * 144);
  convert(WVP, WVPb, (size_t)8 * 384 * 288);
  convert(WB,  WBb,  (size_t)8 * 128 * 12);
  convert(WO,  WOb,  (size_t)8 * 2112 * 384);
  convert(WS,  WSb,  (size_t)384 * 384);

  auto gemm = [&](const bf16_t* A, long long lda, long long bA,
                  const bf16_t* B, long long ldb, long long bB, int btr,
                  float* Cf, bf16_t* Cb, long long ldc, long long bC,
                  const float* bias, const float* res, long long ldres, long long bRes,
                  int M, int N, int K, int relu, int batch) {
    GemmArgs g;
    g.A = A; g.lda = lda; g.batchA = bA;
    g.B = B; g.ldb = ldb; g.batchB = bB; g.btr = btr;
    g.Cf = Cf; g.Cb = Cb; g.ldc = ldc; g.batchC = bC;
    g.bias = bias; g.residual = res; g.ldres = ldres; g.batchRes = bRes;
    g.M = M; g.N = N; g.K = K; g.relu = relu;
    dim3 grid((N + 63) / 64, (M + 63) / 64, batch);
    gemm_bf16_wmma<<<grid, dim3(128), 0, stream>>>(g);
  };

  // --- prologue ---
  ln_rows<<<768, 128, 0, stream>>>(SINGLE, SLNG, SLNB, nullptr, Sh0, 384);
  ln_rows<<<768 * 768, 128, 0, stream>>>(PAIR, PLNG, PLNB, nullptr, pairb, 128);
  init_frames<<<768, 32, 0, stream>>>(Rbuf, tbuf);

  // single = LN(single) @ ws + bs
  gemm(Sh0, 384, 0, WSb, 384, 0, 0, Sf, Sbf, 384, 0, BSv, nullptr, 0, 0,
       768, 384, 384, 0, 1);

  for (int i = 0; i < 8; ++i) {
    const bf16_t* w1_i  = W1b  + (size_t)i * 384 * 384;
    const bf16_t* w2_i  = W2b  + (size_t)i * 384 * 384;
    const bf16_t* w3_i  = W3b  + (size_t)i * 384 * 384;
    const bf16_t* wq_i  = WQb  + (size_t)i * 384 * 192;
    const bf16_t* wk_i  = WKb  + (size_t)i * 384 * 192;
    const bf16_t* wv_i  = WVb  + (size_t)i * 384 * 192;
    const bf16_t* wqp_i = WQPb + (size_t)i * 384 * 144;
    const bf16_t* wkp_i = WKPb + (size_t)i * 384 * 144;
    const bf16_t* wvp_i = WVPb + (size_t)i * 384 * 288;
    const bf16_t* wb_i  = WBb  + (size_t)i * 128 * 12;
    const bf16_t* wo_i  = WOb  + (size_t)i * 2112 * 384;
    const float* wbu_i = WBU + (size_t)i * 384 * 6;
    const float* b1_i  = B1 + i * 384, *b2_i = B2 + i * 384, *b3_i = B3 + i * 384;
    const float* bo_i  = BO + i * 384, *bbu_i = BBU + i * 6;
    const float* g_i   = GAMMA + i * 12;
    const float* l1g = LN1G + i * 384, *l1b = LN1B + i * 384;
    const float* l2g = LN2G + i * 384, *l2b = LN2B + i * 384;

    // q,k,v and point projections
    gemm(Sbf, 384, 0, wq_i,  192, 0, 0, qf,  nullptr, 192, 0, nullptr, nullptr, 0, 0, 768, 192, 384, 0, 1);
    gemm(Sbf, 384, 0, wk_i,  192, 0, 0, kf,  nullptr, 192, 0, nullptr, nullptr, 0, 0, 768, 192, 384, 0, 1);
    gemm(Sbf, 384, 0, wv_i,  192, 0, 0, vf,  nullptr, 192, 0, nullptr, nullptr, 0, 0, 768, 192, 384, 0, 1);
    gemm(Sbf, 384, 0, wqp_i, 144, 0, 0, qpf, nullptr, 144, 0, nullptr, nullptr, 0, 0, 768, 144, 384, 0, 1);
    gemm(Sbf, 384, 0, wkp_i, 144, 0, 0, kpf, nullptr, 144, 0, nullptr, nullptr, 0, 0, 768, 144, 384, 0, 1);
    gemm(Sbf, 384, 0, wvp_i, 288, 0, 0, vpf, nullptr, 288, 0, nullptr, nullptr, 0, 0, 768, 288, 384, 0, 1);

    // pair bias: [589824 x 128] @ [128 x 12] -> biasbuf (ldc 16)
    gemm(pairb, 128, 0, wb_i, 12, 0, 0, biasbuf, nullptr, 16, 0, nullptr, nullptr, 0, 0,
         768 * 768, 12, 128, 0, 1);

    build_attn_inputs<<<dim3(768, 12), 32, 0, stream>>>(
        qf, kf, vf, qpf, kpf, vpf, Rbuf, tbuf, g_i, Abuf, Bbuf, Vcat, rowterm, colterm);

    // logits: per-head [768x32] @ [768x32]^T
    gemm(Abuf, 32, 768 * 32, Bbuf, 32, 768 * 32, 1, logits, nullptr, 768, 589824,
         nullptr, nullptr, 0, 0, 768, 768, 32, 0, 12);

    softmax_rows<<<dim3(768, 12), 256, 0, stream>>>(logits, biasbuf, rowterm, colterm, attnb);

    // Ocat: per-head attn [768x768] @ Vcat_h [768x40]
    gemm(attnb, 768, 589824, Vcat, 480, 40, 0, Ocat, nullptr, 480, 40,
         nullptr, nullptr, 0, 0, 768, 40, 768, 0, 12);

    // o_pair: per-n attn[:,n,:] [12x768] @ pair[n] [768x128]
    gemm(attnb, 589824, 768, pairb, 128, 98304, 0, opair, nullptr, 128, 1536,
         nullptr, nullptr, 0, 0, 12, 128, 768, 0, 768);

    build_cat<<<768, 256, 0, stream>>>(Ocat, opair, Rbuf, tbuf, cat);

    // single = single + cat @ wo + bo
    gemm(cat, 2112, 0, wo_i, 384, 0, 0, Xf, nullptr, 384, 0, bo_i, Sf, 384, 0,
         768, 384, 2112, 0, 1);

    ln_rows<<<768, 128, 0, stream>>>(Xf, l1g, l1b, Yf, Ybf, 384);

    gemm(Ybf, 384, 0, w1_i, 384, 0, 0, nullptr, h1, 384, 0, b1_i, nullptr, 0, 0, 768, 384, 384, 1, 1);
    gemm(h1,  384, 0, w2_i, 384, 0, 0, nullptr, h2, 384, 0, b2_i, nullptr, 0, 0, 768, 384, 384, 1, 1);
    gemm(h2,  384, 0, w3_i, 384, 0, 0, Zf, nullptr, 384, 0, b3_i, Yf, 384, 0, 768, 384, 384, 0, 1);

    ln_rows<<<768, 128, 0, stream>>>(Zf, l2g, l2b, Sf, Sbf, 384);

    frame_update<<<768, 64, 0, stream>>>(Sf, wbu_i, bbu_i, Rbuf, tbuf);
  }

  final_coords<<<(768 * 3 + 255) / 256, 256, 0, stream>>>(LOCAL, Rbuf, tbuf, (float*)d_out);
}
